// TSModel_4896262718051
// MI455X (gfx1250) — matrix operand order repaired and checked by reading, not compiled, not run on hardware
//
#include <hip/hip_runtime.h>

#define U      64
#define TSTEPS 1024
#define NBATCH 2048
#define XC     256      // input timestep chunk resident in LDS
#define HSTR   72       // padded row stride (halves) for h/c LDS tiles

typedef __attribute__((ext_vector_type(16))) _Float16 v16h;
typedef __attribute__((ext_vector_type(8)))  _Float16 v8h;
typedef __attribute__((ext_vector_type(8)))  float    v8f;

// Fast activations: single v_exp_f32 + single v_rcp_f32 (no IEEE div expansion).
__device__ __forceinline__ float sigm(float x) {
  return __builtin_amdgcn_rcpf(1.0f + __expf(-x));
}
__device__ __forceinline__ float tanh_fast(float x) {
  return 2.0f * __builtin_amdgcn_rcpf(1.0f + __expf(-2.0f * x)) - 1.0f;
}

// Load one 16x32 f16 A-fragment for this lane from an LDS tile row.
// ISA layout: lanes 0-15 (row M=lane)  -> K = kb+0..7  and kb+16..23
//             lanes 16-31 (row M=l-16) -> K = kb+8..15 and kb+24..31
// caller passes p = row*HSTR + kbase + 8*hi  (both 16B chunks contiguous)
__device__ __forceinline__ v16h load_frag(const _Float16* p) {
  v8h lo = *(const v8h*)p;
  v8h hh = *(const v8h*)(p + 16);
  v16h r;
#pragma unroll
  for (int i = 0; i < 8; ++i) { r[i] = lo[i]; r[i + 8] = hh[i]; }
  return r;
}

__global__ __launch_bounds__(128, 1)
void lstm2_wmma_kernel(const float* __restrict__ input,
                       const float* __restrict__ W_ih0, const float* __restrict__ W_hh0,
                       const float* __restrict__ b_ih0, const float* __restrict__ b_hh0,
                       const float* __restrict__ W_ih1, const float* __restrict__ W_hh1,
                       const float* __restrict__ b_ih1, const float* __restrict__ b_hh1,
                       float* __restrict__ out)
{
  __shared__ __align__(16) float    xin[16 * XC];        // 16 KB input chunk
  __shared__ __align__(16) _Float16 hbuf[2][16 * HSTR];  // double-buffered h0 (f16)
  __shared__ __align__(16) _Float16 cbuf[2][16 * HSTR];  // double-buffered c0 (f16)
  __shared__ __align__(16) float    g1buf[64];           // layer-1 gate transpose

  const int tid  = threadIdx.x;
  const int wv   = tid >> 5;      // wave id 0..3
  const int lane = tid & 31;
  const int hi   = lane >> 4;     // lane half
  const int lrow = lane & 15;
  const int row0 = blockIdx.x * 16;

  // ---- resident layer-0 recurrent weights: B[k][n] = W_hh0[n][k], f16 ----
  // wave wv owns gate columns n = g*64 + 16*wv + lrow for g in {i,f,g,o}
  v16h Bf[4][2];
  float bias0[4], wih0[4];
#pragma unroll
  for (int g = 0; g < 4; ++g) {
    const int n = g * 64 + 16 * wv + lrow;
    bias0[g] = b_ih0[n] + b_hh0[n];
    wih0[g]  = W_ih0[n];                       // W_ih0 is (256,1)
#pragma unroll
    for (int kb = 0; kb < 2; ++kb) {
      const float* src = W_hh0 + n * U + kb * 32 + 8 * hi;
      v16h b;
#pragma unroll
      for (int j = 0; j < 16; ++j) b[j] = (_Float16)src[j + (j & 8)];
      Bf[g][kb] = b;
    }
  }

  // ---- layer-1 input weights as 64x16 B-fragments (cols >= 4 zero) ----
  v16h B1[2];
  float bias1 = 0.0f, whh1 = 0.0f;
  if (lrow < 4) { bias1 = b_ih1[lrow] + b_hh1[lrow]; whh1 = W_hh1[lrow]; }
#pragma unroll
  for (int kb = 0; kb < 2; ++kb) {
    v16h b;
#pragma unroll
    for (int j = 0; j < 16; ++j) {
      const int k = kb * 32 + 8 * hi + j + (j & 8);
      b[j] = (lrow < 4) ? (_Float16)W_ih1[lrow * U + k] : (_Float16)0.0f;
    }
    B1[kb] = b;
  }

  // h0(t=0) = 0
  for (int i = tid; i < 16 * HSTR; i += 128) hbuf[0][i] = (_Float16)0.0f;

  float cst[8];
#pragma unroll
  for (int r = 0; r < 8; ++r) cst[r] = 0.0f;
  float h1 = 0.0f, c1 = 0.0f;

  const int arow = lrow * HSTR;

  for (int t = 0; t < TSTEPS; ++t) {
    const int cur = t & 1, nxt = cur ^ 1;

    if ((t & (XC - 1)) == 0) {           // refresh input chunk
      __syncthreads();
      for (int i = tid; i < 16 * XC; i += 128) {
        const int m = i >> 8, j = i & (XC - 1);
        xin[i] = input[(row0 + m) * TSTEPS + t + j];
      }
      __syncthreads();
    }

    // ---- layer 0: gates = h0 @ Whh0^T + x*w_ih0 + bias ----
    v16h a0 = load_frag(&hbuf[cur][arow + 0  + 8 * hi]);
    v16h a1 = load_frag(&hbuf[cur][arow + 32 + 8 * hi]);

    float xv[8];
    const int tc = t & (XC - 1);
#pragma unroll
    for (int r = 0; r < 8; ++r) xv[r] = xin[(r + 8 * hi) * XC + tc];  // LDS broadcast

    v8f acc[4];
#pragma unroll
    for (int g = 0; g < 4; ++g) {
      v8f c;
#pragma unroll
      for (int r = 0; r < 8; ++r) c[r] = bias0[g] + xv[r] * wih0[g];
      c = __builtin_amdgcn_wmma_f32_16x16x32_f16(false, a0, false, Bf[g][0],
                                                 (short)0, c, false, false);
      c = __builtin_amdgcn_wmma_f32_16x16x32_f16(false, a1, false, Bf[g][1],
                                                 (short)0, c, false, false);
      acc[g] = c;
    }

    // ---- pointwise LSTM update; stash h/c as f16 for next step & layer 1 ----
#pragma unroll
    for (int r = 0; r < 8; ++r) {
      const float ig = sigm(acc[0][r]);
      const float fg = sigm(acc[1][r]);
      const float gg = tanh_fast(acc[2][r]);
      const float og = sigm(acc[3][r]);
      const float cn = fg * cst[r] + ig * gg;
      const float hn = og * tanh_fast(cn);
      cst[r] = cn;
      const int idx = (r + 8 * hi) * HSTR + 16 * wv + lrow;  // (row, unit)
      hbuf[nxt][idx] = (_Float16)hn;
      cbuf[nxt][idx] = (_Float16)cn;
    }

    __syncthreads();   // single per-step barrier (h/c double-buffered)

    // ---- layer 1 on wave 0: gates1 = c0n @ Wih1^T + h1*whh1 + bias1 ----
    if (wv == 0) {
      v16h ca0 = load_frag(&cbuf[nxt][arow + 0  + 8 * hi]);
      v16h ca1 = load_frag(&cbuf[nxt][arow + 32 + 8 * hi]);
      v8f g1;
#pragma unroll
      for (int r = 0; r < 8; ++r)
        g1[r] = bias1 + __shfl(h1, r + 8 * hi, 32) * whh1;
      g1 = __builtin_amdgcn_wmma_f32_16x16x32_f16(false, ca0, false, B1[0],
                                                  (short)0, g1, false, false);
      g1 = __builtin_amdgcn_wmma_f32_16x16x32_f16(false, ca1, false, B1[1],
                                                  (short)0, g1, false, false);
      if (lrow < 4) {            // transpose 16x4 gate tile through LDS
#pragma unroll
        for (int r = 0; r < 8; ++r) g1buf[(r + 8 * hi) * 4 + lrow] = g1[r];
      }
      asm volatile("s_wait_dscnt 0x0" ::: "memory");  // same-wave LDS RAW
      if (lane < 16) {
        const float4 gv = *(const float4*)&g1buf[lane * 4];
        const float ig = sigm(gv.x), fg = sigm(gv.y);
        const float gg = tanh_fast(gv.z), og = sigm(gv.w);
        const float c1n = fg * c1 + ig * gg;
        c1 = c1n;
        h1 = og * tanh_fast(c1n);
        out[(row0 + lane) * TSTEPS + t] = c1n;   // y_t = c1
      }
    }
  }
}

extern "C" void kernel_launch(void* const* d_in, const int* in_sizes, int n_in,
                              void* d_out, int out_size, void* d_ws, size_t ws_size,
                              hipStream_t stream) {
  (void)in_sizes; (void)n_in; (void)out_size; (void)d_ws; (void)ws_size;
  lstm2_wmma_kernel<<<NBATCH / 16, 128, 0, stream>>>(
      (const float*)d_in[0],
      (const float*)d_in[1], (const float*)d_in[2],
      (const float*)d_in[3], (const float*)d_in[4],
      (const float*)d_in[5], (const float*)d_in[6],
      (const float*)d_in[7], (const float*)d_in[8],
      (float*)d_out);
}
